// AttnBlock_24696061952213
// MI455X (gfx1250) — matrix operand order repaired and checked by reading, not compiled, not run on hardware
//
#include <hip/hip_runtime.h>
#include <cstdint>
#include <cstddef>

// ---------------------------------------------------------------------------
// DiT block for MI455X (gfx1250, wave32, WMMA + TDM).  bf16 WMMA, fp32 accum.
// All LDS tile staging (GEMM A/B, attention Q/K/V) goes through the Tensor
// Data Mover when available: zero-VGPR DMA, double-buffered, one barrier per
// step.  Fallback path uses direct global->LDS copies (no long-lived regs).
// ---------------------------------------------------------------------------

typedef __bf16 bf16;
typedef __attribute__((ext_vector_type(16))) bf16  v16bf;
typedef __attribute__((ext_vector_type(8)))  float v8f;
typedef __attribute__((ext_vector_type(4)))  unsigned int u32x4;
typedef __attribute__((ext_vector_type(8)))  int i32x8;
typedef __attribute__((ext_vector_type(4)))  int i32x4;

#define WMMA_BF16(a, b, c) \
  __builtin_amdgcn_wmma_f32_16x16x32_bf16(false, (a), false, (b), (short)0, (c), false, false)

#if defined(__AMDGCN__) && __has_builtin(__builtin_amdgcn_tensor_load_to_lds) && \
    __has_builtin(__builtin_amdgcn_s_wait_tensorcnt)
#define USE_TDM 1
#else
#define USE_TDM 0
#endif

union Frag16 { v16bf v; bf16 h[16]; uint4 q[2]; };

static constexpr int B_  = 4;
static constexpr int N_  = 1024;
static constexpr int D_  = 1024;
static constexpr int H_  = 16;
static constexpr int DH_ = 64;
static constexpr int M_  = B_ * N_;     // 4096 tokens
static constexpr int MLP_ = 4 * D_;     // 4096

// ------------------------- small helpers -----------------------------------

__device__ __forceinline__ float hred_max(float v) {   // reduce over 16-lane half
  for (int m = 8; m >= 1; m >>= 1) v = fmaxf(v, __shfl_xor(v, m, 32));
  return v;
}
__device__ __forceinline__ float hred_sum(float v) {
  for (int m = 8; m >= 1; m >>= 1) v += __shfl_xor(v, m, 32);
  return v;
}
__device__ __forceinline__ float gelu_tanh(float x) {
  float x3 = x * x * x;
  return 0.5f * x * (1.f + tanhf(0.7978845608028654f * (x + 0.044715f * x3)));
}

#if USE_TDM
// Issue a TDM 2D tile load (bf16 elements): global[tile_h][tile_w] with row
// stride row_stride_elems -> LDS (row-major, packed).  D# layout per CDNA5
// ISA ch.8 (group0: count/lds/global/type, group1: dsize/dims/stride).
__device__ __forceinline__ void tdm_load_tile(uint32_t lds_off, const bf16* gptr,
                                              uint32_t tile_w, uint32_t tile_h,
                                              uint32_t row_stride_elems) {
  uint64_t ga = (uint64_t)(uintptr_t)gptr;
  u32x4 g0;
  g0[0] = 1u;                                                  // count=1 (user D#)
  g0[1] = lds_off;                                             // lds_addr (bytes)
  g0[2] = (uint32_t)ga;                                        // global_addr[31:0]
  g0[3] = (uint32_t)((ga >> 32) & 0x01FFFFFFu) | (2u << 30);   // addr[56:32]|type=2
  i32x8 g1;
  g1[0] = (int)(1u << 16);                                     // data_size=1 (2B)
  g1[1] = (int)((tile_w & 0xFFFFu) << 16);                     // tensor_dim0 lo
  g1[2] = (int)(((tile_w >> 16) & 0xFFFFu) |                   // tensor_dim0 hi
                ((tile_h & 0xFFFFu) << 16));                   // tensor_dim1 lo
  g1[3] = (int)(((tile_h >> 16) & 0xFFFFu) |                   // tensor_dim1 hi
                ((tile_w & 0xFFFFu) << 16));                   // tile_dim0
  g1[4] = (int)(tile_h & 0xFFFFu);                             // tile_dim1 (dim2=0)
  g1[5] = (int)row_stride_elems;                               // dim0_stride[31:0]
  g1[6] = 0;                                                   // stride hi / d1s lo
  g1[7] = 0;
  i32x4 gz = {0, 0, 0, 0};
#if defined(__clang_major__) && (__clang_major__ >= 23)
  i32x8 gz8 = {0, 0, 0, 0, 0, 0, 0, 0};
  __builtin_amdgcn_tensor_load_to_lds(g0, g1, gz, gz, gz8, 0);
#else
  __builtin_amdgcn_tensor_load_to_lds(g0, g1, gz, gz, 0);
#endif
}
#endif

// ------------------------- trivial kernels ---------------------------------

__global__ __launch_bounds__(256) void silu_kernel(const float* __restrict__ c,
                                                   float* __restrict__ out, int n) {
  int i = blockIdx.x * 256 + threadIdx.x;
  if (i < n) { float x = c[i]; out[i] = x / (1.f + __expf(-x)); }
}

// fp32 [K][N] -> bf16 [N][K] tiled transpose (one-time per launch).
__global__ __launch_bounds__(256) void transpose_cvt_kernel(const float* __restrict__ W,
                                                            bf16* __restrict__ Wt,
                                                            int K, int N) {
  __shared__ float t[32][33];
  int n0 = blockIdx.x * 32, k0 = blockIdx.y * 32;
  int c = threadIdx.x & 31, r8 = threadIdx.x >> 5;
#pragma unroll
  for (int i = 0; i < 4; ++i) {
    int r = r8 + i * 8;
    t[r][c] = W[(size_t)(k0 + r) * N + n0 + c];
  }
  __syncthreads();
#pragma unroll
  for (int i = 0; i < 4; ++i) {
    int r = r8 + i * 8;
    Wt[(size_t)(n0 + r) * K + k0 + c] = (bf16)t[c][r];
  }
}

// mod[b][n] = silu(c[b]) . Wada[:,n] + bada[n]; each thread does all 4 batches
// so Wada is read exactly once (24 MB).  silu(c) staged in LDS.
__global__ __launch_bounds__(256) void ada_kernel(const float* __restrict__ sc,
                                                  const float* __restrict__ Wada,
                                                  const float* __restrict__ bada,
                                                  float* __restrict__ mod) {
  __shared__ float scs[B_ * D_];
  int tid = threadIdx.x;
#pragma unroll
  for (int i = 0; i < (B_ * D_) / 256; ++i) scs[tid + i * 256] = sc[tid + i * 256];
  __syncthreads();
  int n = blockIdx.x * 256 + tid;
  float bv = bada[n];
  float a0 = bv, a1 = bv, a2 = bv, a3 = bv;
  for (int k = 0; k < D_; ++k) {
    float w = Wada[(size_t)k * 6 * D_ + n];
    a0 += scs[k] * w;
    a1 += scs[D_ + k] * w;
    a2 += scs[2 * D_ + k] * w;
    a3 += scs[3 * D_ + k] * w;
  }
  mod[n]           = a0;
  mod[6 * D_ + n]  = a1;
  mod[12 * D_ + n] = a2;
  mod[18 * D_ + n] = a3;
}

// LayerNorm + adaLN modulate -> bf16.  One block per token, 256 threads.
__global__ __launch_bounds__(256) void ln_mod_kernel(const float* __restrict__ X,
                                                     const float* __restrict__ mod,
                                                     int shift_seg, int scale_seg,
                                                     bf16* __restrict__ Hout) {
  int t = blockIdx.x, tid = threadIdx.x;
  int b = t >> 10;
  const float* x = X + (size_t)t * D_;
  float v[4], s = 0.f, s2 = 0.f;
#pragma unroll
  for (int i = 0; i < 4; ++i) { v[i] = x[tid + i * 256]; s += v[i]; s2 += v[i] * v[i]; }
  for (int m = 16; m >= 1; m >>= 1) { s += __shfl_xor(s, m, 32); s2 += __shfl_xor(s2, m, 32); }
  __shared__ float red[2][8];
  int wid = tid >> 5, lane = tid & 31;
  if (lane == 0) { red[0][wid] = s; red[1][wid] = s2; }
  __syncthreads();
  if (tid == 0) {
    float a = 0.f, q = 0.f;
    for (int w = 0; w < 8; ++w) { a += red[0][w]; q += red[1][w]; }
    red[0][0] = a; red[1][0] = q;
  }
  __syncthreads();
  float mu   = red[0][0] * (1.f / D_);
  float var  = red[1][0] * (1.f / D_) - mu * mu;
  float rstd = rsqrtf(var + 1e-6f);
  const float* sh  = mod + (size_t)b * 6 * D_ + (size_t)shift_seg * D_;
  const float* scl = mod + (size_t)b * 6 * D_ + (size_t)scale_seg * D_;
  bf16* h = Hout + (size_t)t * D_;
#pragma unroll
  for (int i = 0; i < 4; ++i) {
    int cidx = tid + i * 256;
    float y = (v[i] - mu) * rstd * (1.f + scl[cidx]) + sh[cidx];
    h[cidx] = (bf16)y;
  }
}

// ------------------------- tiled WMMA GEMM ---------------------------------
// C[M,Nc] = A[M,K] (bf16) @ Wt[Nc,K]^T (bf16, pre-transposed weights).
//   EPI=0: bf16 out = acc + bias[n]
//   EPI=1: bf16 out = gelu(acc + bias[n])
//   EPI=2: f32  out = res[m][n] + gate[m/1024][n] * (acc + bias[n])
//   EPI=3: qkv split: col<1024 -> k buf [tok][1024]; col>=1024 -> v transposed
// Block tile 128x128, K-step 64, 256 threads (8 waves), wave tile 32x64.
// Ping-pong LDS; one barrier per K-step.

template <int EPI>
__global__ __launch_bounds__(256) void gemm_bf16_kernel(
    const bf16* __restrict__ A, const bf16* __restrict__ Wt,
    const float* __restrict__ bias, int M, int Nc, int K,
    bf16* __restrict__ outb, bf16* __restrict__ outb2, float* __restrict__ outf,
    const float* __restrict__ res, const float* __restrict__ gate) {
  __shared__ __align__(16) bf16 As[2][128 * 64];
  __shared__ __align__(16) bf16 Bt[2][128 * 64];   // Bt[n][k]

  const int tid = threadIdx.x;
  const int wid = tid >> 5, lane = tid & 31, lm = lane & 15, g = lane >> 4;
  const int wm = wid & 3, wn = wid >> 2;        // wave tile: rows wm*32, cols wn*64
  const int m0 = blockIdx.y * 128, n0 = blockIdx.x * 128;
  const int nsteps = K / 64;

  v8f acc[2][4] = {};

  // ---- prologue: stage tile 0 into buffer 0 ----
#if USE_TDM
  if (tid == 0)
    tdm_load_tile((uint32_t)(uintptr_t)&As[0][0], A + (size_t)m0 * K, 64, 128, (uint32_t)K);
  else if (tid == 32)
    tdm_load_tile((uint32_t)(uintptr_t)&Bt[0][0], Wt + (size_t)n0 * K, 64, 128, (uint32_t)K);
  if (tid == 0 || tid == 32) __builtin_amdgcn_s_wait_tensorcnt(0);
#else
#pragma unroll
  for (int t = 0; t < 4; ++t) {                 // 128x64 bf16 = 1024 uint4 each
    int idx = tid + t * 256;
    int row = idx >> 3, c8 = idx & 7;
    *(uint4*)(As[0] + row * 64 + c8 * 8) =
        *(const uint4*)(A + (size_t)(m0 + row) * K + c8 * 8);
    *(uint4*)(Bt[0] + row * 64 + c8 * 8) =
        *(const uint4*)(Wt + (size_t)(n0 + row) * K + c8 * 8);
  }
#endif
  __syncthreads();

  int cur = 0;
  for (int s = 0; s < nsteps; ++s) {
    const bool pre = (s + 1 < nsteps);
    const int k1 = (s + 1) * 64;

    const bf16* Ac = As[cur];
    const bf16* Bc = Bt[cur];
    auto compute_kk = [&](int kk) {
      Frag16 af[2], bfg[4];
#pragma unroll
      for (int i = 0; i < 2; ++i) {
        const bf16* ar = Ac + (wm * 32 + i * 16 + lm) * 64 + kk;
        af[i].q[0] = *(const uint4*)(ar + g * 8);
        af[i].q[1] = *(const uint4*)(ar + 16 + g * 8);
      }
#pragma unroll
      for (int j = 0; j < 4; ++j) {
        const bf16* br = Bc + (wn * 64 + j * 16 + lm) * 64 + kk;
        bfg[j].q[0] = *(const uint4*)(br + g * 8);
        bfg[j].q[1] = *(const uint4*)(br + 16 + g * 8);
      }
#pragma unroll
      for (int i = 0; i < 2; ++i)
#pragma unroll
        for (int j = 0; j < 4; ++j)
          acc[i][j] = WMMA_BF16(af[i].v, bfg[j].v, acc[i][j]);
    };

#if USE_TDM
    if (pre) {                                  // DMA next tile while computing
      if (tid == 0)
        tdm_load_tile((uint32_t)(uintptr_t)&As[cur ^ 1][0],
                      A + (size_t)m0 * K + k1, 64, 128, (uint32_t)K);
      else if (tid == 32)
        tdm_load_tile((uint32_t)(uintptr_t)&Bt[cur ^ 1][0],
                      Wt + (size_t)n0 * K + k1, 64, 128, (uint32_t)K);
    }
    compute_kk(0);
    compute_kk(32);
    if (pre && (tid == 0 || tid == 32)) __builtin_amdgcn_s_wait_tensorcnt(0);
#else
    if (pre) {                                  // direct global->LDS commit
#pragma unroll
      for (int t = 0; t < 4; ++t) {
        int idx = tid + t * 256;
        int row = idx >> 3, c8 = idx & 7;
        *(uint4*)(As[cur ^ 1] + row * 64 + c8 * 8) =
            *(const uint4*)(A + (size_t)(m0 + row) * K + k1 + c8 * 8);
        *(uint4*)(Bt[cur ^ 1] + row * 64 + c8 * 8) =
            *(const uint4*)(Wt + (size_t)(n0 + row) * K + k1 + c8 * 8);
      }
    }
    compute_kk(0);
    compute_kk(32);
#endif
    __syncthreads();                            // retire cur / release cur^1
    cur ^= 1;
  }

  // epilogue: lane owns rows (g*8 + r) of each 16x16 tile, col = lm
#pragma unroll
  for (int i = 0; i < 2; ++i)
#pragma unroll
    for (int j = 0; j < 4; ++j) {
      int gcol = n0 + wn * 64 + j * 16 + lm;
      float bv = bias[gcol];
#pragma unroll
      for (int r = 0; r < 8; ++r) {
        int grow = m0 + wm * 32 + i * 16 + g * 8 + r;
        float v = acc[i][j][r] + bv;
        if (EPI == 0) {
          outb[(size_t)grow * Nc + gcol] = (bf16)v;
        } else if (EPI == 1) {
          outb[(size_t)grow * Nc + gcol] = (bf16)gelu_tanh(v);
        } else if (EPI == 2) {
          int bb = grow >> 10;
          outf[(size_t)grow * Nc + gcol] =
              res[(size_t)grow * Nc + gcol] + gate[(size_t)bb * 6 * D_ + gcol] * v;
        } else {                                  // EPI == 3 : qkv split
          if (gcol < D_) {
            outb[(size_t)grow * D_ + gcol] = (bf16)v;          // K, [tok][D]
          } else {
            int hd = (gcol - D_) >> 6, d = (gcol - D_) & 63;
            int bb = grow >> 10, nn = grow & (N_ - 1);
            outb2[(((size_t)bb * H_ + hd) * DH_ + d) * N_ + nn] = (bf16)v;  // V^T
          }
        }
      }
    }
}

// ------------------------- flash attention ---------------------------------
// grid = (N/64, H, B), 128 threads (4 waves). Each wave owns 16 q-rows.
// Q:[4096,1024] bf16 ; Kb:[4096,1024] bf16 ; Vt:[B*H, 64, 1024] bf16 (d-major)
// O:[4096,1024] bf16.  scale = DH^-0.5 = 0.125.  Q/K/V tiles staged by TDM
// (zero-VGPR DMA, double-buffered K/V, one barrier per tile); fallback path
// uses direct global->LDS copies so nothing is held in registers.

__global__ __launch_bounds__(128)
void attn_kernel(const bf16* __restrict__ Q,
                 const bf16* __restrict__ Kb,
                 const bf16* __restrict__ Vt,
                 bf16* __restrict__ O) {
  const int m0 = blockIdx.x * 64, head = blockIdx.y, b = blockIdx.z;
  const int tid = threadIdx.x;
  const int wid = tid >> 5, lane = tid & 31, lm = lane & 15, g = lane >> 4;

  __shared__ __align__(16) bf16 qs[64 * 64];
  __shared__ __align__(16) bf16 ks[2][32 * 64];     // ks[key][d]
  __shared__ __align__(16) bf16 vt[2][64 * 32];     // vt[d][key]
  __shared__ __align__(16) bf16 ps[4][16 * 32];     // per-wave P staging

  const bf16* qbase = Q + ((size_t)(b * N_ + m0)) * D_ + head * DH_;
  const bf16* kbase = Kb + ((size_t)(b * N_)) * D_ + head * DH_;
  const bf16* vbase = Vt + ((size_t)(b * H_ + head) * DH_) * N_;

  // ---- prologue: stage q tile + K/V tile 0 ----
#if USE_TDM
  if (tid == 0)
    tdm_load_tile((uint32_t)(uintptr_t)qs,    qbase, 64, 64, (uint32_t)D_);
  else if (tid == 32)
    tdm_load_tile((uint32_t)(uintptr_t)ks[0], kbase, 64, 32, (uint32_t)D_);
  else if (tid == 64)
    tdm_load_tile((uint32_t)(uintptr_t)vt[0], vbase, 32, 64, (uint32_t)N_);
  if (tid == 0 || tid == 32 || tid == 64) __builtin_amdgcn_s_wait_tensorcnt(0);
#else
#pragma unroll
  for (int t = 0; t < 4; ++t) {                     // q tile 64x64 = 512 uint4
    int idx = tid + t * 128;
    int row = idx >> 3, c8 = idx & 7;
    *(uint4*)(qs + row * 64 + c8 * 8) = *(const uint4*)(qbase + (size_t)row * D_ + c8 * 8);
  }
#pragma unroll
  for (int t = 0; t < 2; ++t) {                     // K tile 0: 32x64 = 256 uint4
    int idx = tid + t * 128;
    int row = idx >> 3, c8 = idx & 7;
    *(uint4*)(ks[0] + row * 64 + c8 * 8) =
        *(const uint4*)(kbase + (size_t)row * D_ + c8 * 8);
  }
#pragma unroll
  for (int t = 0; t < 2; ++t) {                     // V^T tile 0: 64x32 = 256 uint4
    int idx = tid + t * 128;
    int row = idx >> 2, c4 = idx & 3;
    *(uint4*)(vt[0] + row * 32 + c4 * 8) =
        *(const uint4*)(vbase + (size_t)row * N_ + c4 * 8);
  }
#endif
  __syncthreads();

  v8f o[4] = {};
  float mrun[8], lrun[8];
#pragma unroll
  for (int r = 0; r < 8; ++r) { mrun[r] = -3.0e38f; lrun[r] = 0.f; }

  int cur = 0;
  for (int kt = 0; kt < N_ / 32; ++kt) {
    const bool pre = (kt + 1 < N_ / 32);

#if USE_TDM
    if (pre) {                                      // DMA next K/V tiles
      if (tid == 0)
        tdm_load_tile((uint32_t)(uintptr_t)ks[cur ^ 1],
                      kbase + (size_t)(kt + 1) * 32 * D_, 64, 32, (uint32_t)D_);
      else if (tid == 32)
        tdm_load_tile((uint32_t)(uintptr_t)vt[cur ^ 1],
                      vbase + (kt + 1) * 32, 32, 64, (uint32_t)N_);
    }
#else
    if (pre) {                                      // direct global->LDS commit
#pragma unroll
      for (int t = 0; t < 2; ++t) {
        int idx = tid + t * 128;
        int krow = idx >> 3, c8 = idx & 7;
        *(uint4*)(ks[cur ^ 1] + krow * 64 + c8 * 8) =
            *(const uint4*)(kbase + (size_t)((kt + 1) * 32 + krow) * D_ + c8 * 8);
        int vrow = idx >> 2, c4 = idx & 3;
        *(uint4*)(vt[cur ^ 1] + vrow * 32 + c4 * 8) =
            *(const uint4*)(vbase + (size_t)vrow * N_ + (kt + 1) * 32 + c4 * 8);
      }
    }
#endif

    // S = q_sub[16x64] @ k_tile^T  -> two 16x16 accumulators (keys 0-15,16-31)
    v8f s0 = {}, s1 = {};
#pragma unroll
    for (int kk = 0; kk < 2; ++kk) {
      Frag16 af;
      const bf16* ar = qs + (wid * 16 + lm) * 64 + kk * 32;
      af.q[0] = *(const uint4*)(ar + g * 8);
      af.q[1] = *(const uint4*)(ar + 16 + g * 8);
      Frag16 b0, b1;
      const bf16* br0 = ks[cur] + lm * 64 + kk * 32;
      const bf16* br1 = ks[cur] + (16 + lm) * 64 + kk * 32;
      b0.q[0] = *(const uint4*)(br0 + g * 8); b0.q[1] = *(const uint4*)(br0 + 16 + g * 8);
      b1.q[0] = *(const uint4*)(br1 + g * 8); b1.q[1] = *(const uint4*)(br1 + 16 + g * 8);
      s0 = WMMA_BF16(af.v, b0.v, s0);
      s1 = WMMA_BF16(af.v, b1.v, s1);
    }

    // streaming softmax update
#pragma unroll
    for (int r = 0; r < 8; ++r) {
      float a = s0[r] * 0.125f, c2 = s1[r] * 0.125f;
      float tm = hred_max(fmaxf(a, c2));
      float nm = fmaxf(mrun[r], tm);
      float al = __expf(mrun[r] - nm);
      float p0 = __expf(a - nm), p1 = __expf(c2 - nm);
      float rs = hred_sum(p0 + p1);
      lrun[r] = lrun[r] * al + rs;
      mrun[r] = nm;
#pragma unroll
      for (int nt = 0; nt < 4; ++nt) o[nt][r] *= al;
      ps[wid][(g * 8 + r) * 32 + lm]      = (bf16)p0;
      ps[wid][(g * 8 + r) * 32 + 16 + lm] = (bf16)p1;
    }

    // O += P[16x32] @ V[32x64]   (P re-read from LDS in A-fragment layout)
    Frag16 ap;
    const bf16* pr = ps[wid] + lm * 32;
    ap.q[0] = *(const uint4*)(pr + g * 8);
    ap.q[1] = *(const uint4*)(pr + 16 + g * 8);
#pragma unroll
    for (int nt = 0; nt < 4; ++nt) {
      Frag16 bv;
      const bf16* vr = vt[cur] + (nt * 16 + lm) * 32;
      bv.q[0] = *(const uint4*)(vr + g * 8);
      bv.q[1] = *(const uint4*)(vr + 16 + g * 8);
      o[nt] = WMMA_BF16(ap.v, bv.v, o[nt]);
    }

#if USE_TDM
    if (pre && (tid == 0 || tid == 32)) __builtin_amdgcn_s_wait_tensorcnt(0);
#endif
    __syncthreads();
    cur ^= 1;
  }

  bf16* obase = O + ((size_t)(b * N_ + m0 + wid * 16)) * D_ + head * DH_;
#pragma unroll
  for (int r = 0; r < 8; ++r) {
    float inv = 1.f / lrun[r];
#pragma unroll
    for (int nt = 0; nt < 4; ++nt)
      obase[(size_t)(g * 8 + r) * D_ + nt * 16 + lm] = (bf16)(o[nt][r] * inv);
  }
}

// ------------------------- host orchestration ------------------------------

extern "C" void kernel_launch(void* const* d_in, const int* in_sizes, int n_in,
                              void* d_out, int out_size, void* d_ws, size_t ws_size,
                              hipStream_t stream) {
  const float* x    = (const float*)d_in[0];
  const float* c    = (const float*)d_in[1];
  const float* Wq   = (const float*)d_in[2];
  const float* bq   = (const float*)d_in[3];
  const float* Wkv  = (const float*)d_in[4];
  const float* bkv  = (const float*)d_in[5];
  const float* Wo   = (const float*)d_in[6];
  const float* bo   = (const float*)d_in[7];
  const float* W1   = (const float*)d_in[8];
  const float* b1   = (const float*)d_in[9];
  const float* W2   = (const float*)d_in[10];
  const float* b2   = (const float*)d_in[11];
  const float* Wada = (const float*)d_in[12];
  const float* bada = (const float*)d_in[13];
  float* out = (float*)d_out;

  char* ws = (char*)d_ws;
  size_t o_wq  = 0;                                     // Wq^T  [D,D]   bf16
  size_t o_wkv = o_wq  + (size_t)D_ * D_ * 2;           // Wkv^T [2D,D]  bf16
  size_t o_wo  = o_wkv + (size_t)D_ * 2 * D_ * 2;       // Wo^T  [D,D]   bf16
  size_t o_w1  = o_wo  + (size_t)D_ * D_ * 2;           // W1^T  [MLP,D] bf16
  size_t o_w2  = o_w1  + (size_t)D_ * MLP_ * 2;         // W2^T  [D,MLP] bf16
  size_t o_sc  = o_w2  + (size_t)MLP_ * D_ * 2;         // silu(c)       f32
  size_t o_mod = o_sc  + (size_t)B_ * D_ * 4;           // mod [B,6D]    f32
  size_t o_h   = o_mod + (size_t)B_ * 6 * D_ * 4;       // h             bf16
  size_t o_q   = o_h   + (size_t)M_ * D_ * 2;           // q   [M,D]     bf16
  size_t o_k   = o_q   + (size_t)M_ * D_ * 2;           // k   [M,D]     bf16
  size_t o_vt  = o_k   + (size_t)M_ * D_ * 2;           // v^T [B*H,64,N]bf16
  size_t o_att = o_vt  + (size_t)M_ * D_ * 2;           // attn out      bf16
  size_t o_x1  = o_att + (size_t)M_ * D_ * 2;           // x after attn  f32
  size_t o_mlp = o_x1  + (size_t)M_ * D_ * 4;           // MLP hidden    bf16
  // total = o_mlp + M_*MLP_*2  (~112 MB)

  bf16*  wq_b  = (bf16*)(ws + o_wq);
  bf16*  wkv_b = (bf16*)(ws + o_wkv);
  bf16*  wo_b  = (bf16*)(ws + o_wo);
  bf16*  w1_b  = (bf16*)(ws + o_w1);
  bf16*  w2_b  = (bf16*)(ws + o_w2);
  float* sc_f  = (float*)(ws + o_sc);
  float* mod_f = (float*)(ws + o_mod);
  bf16*  h_b   = (bf16*)(ws + o_h);
  bf16*  q_b   = (bf16*)(ws + o_q);
  bf16*  k_b   = (bf16*)(ws + o_k);
  bf16*  vt_b  = (bf16*)(ws + o_vt);
  bf16*  att_b = (bf16*)(ws + o_att);
  float* x1_f  = (float*)(ws + o_x1);
  bf16*  mlp_b = (bf16*)(ws + o_mlp);

  auto tcvt = [&](const float* s, bf16* d, int K, int N) {   // [K,N]f32 -> [N,K]bf16
    transpose_cvt_kernel<<<dim3(N / 32, K / 32), 256, 0, stream>>>(s, d, K, N);
  };
  tcvt(Wq,  wq_b,  D_, D_);
  tcvt(Wkv, wkv_b, D_, 2 * D_);
  tcvt(Wo,  wo_b,  D_, D_);
  tcvt(W1,  w1_b,  D_, MLP_);
  tcvt(W2,  w2_b,  MLP_, D_);

  silu_kernel<<<(B_ * D_) / 256, 256, 0, stream>>>(c, sc_f, B_ * D_);
  ada_kernel<<<(6 * D_) / 256, 256, 0, stream>>>(sc_f, Wada, bada, mod_f);

  // --- attention branch ---
  ln_mod_kernel<<<M_, 256, 0, stream>>>(x, mod_f, 0, 1, h_b);   // shift_msa, scale_msa
  gemm_bf16_kernel<0><<<dim3(D_ / 128, M_ / 128), 256, 0, stream>>>(
      h_b, wq_b, bq, M_, D_, D_, q_b, nullptr, nullptr, nullptr, nullptr);
  gemm_bf16_kernel<3><<<dim3(2 * D_ / 128, M_ / 128), 256, 0, stream>>>(
      h_b, wkv_b, bkv, M_, 2 * D_, D_, k_b, vt_b, nullptr, nullptr, nullptr);
  attn_kernel<<<dim3(N_ / 64, H_, B_), 128, 0, stream>>>(q_b, k_b, vt_b, att_b);
  gemm_bf16_kernel<2><<<dim3(D_ / 128, M_ / 128), 256, 0, stream>>>(
      att_b, wo_b, bo, M_, D_, D_, nullptr, nullptr, x1_f, x, mod_f + 2 * D_);

  // --- MLP branch ---
  ln_mod_kernel<<<M_, 256, 0, stream>>>(x1_f, mod_f, 3, 4, h_b); // shift_mlp, scale_mlp
  gemm_bf16_kernel<1><<<dim3(MLP_ / 128, M_ / 128), 256, 0, stream>>>(
      h_b, w1_b, b1, M_, MLP_, D_, mlp_b, nullptr, nullptr, nullptr, nullptr);
  gemm_bf16_kernel<2><<<dim3(D_ / 128, M_ / 128), 256, 0, stream>>>(
      mlp_b, w2_b, b2, M_, D_, MLP_, nullptr, nullptr, out, x1_f, mod_f + 5 * D_);
}
